// HRMNet_20684562498087
// MI455X (gfx1250) — matrix operand order repaired and checked by reading, not compiled, not run on hardware
//
#include <hip/hip_runtime.h>
#include <math.h>
#include <stdint.h>

// ---------------------------------------------------------------------------
// HRM forward for MI455X (gfx1250, wave32).
//  - all matmuls via one batched/strided WMMA GEMM (v_wmma_f32_16x16x32_bf16)
//  - weights converted once/call to bf16 [N,K] (k-contiguous) so B tiles are
//    staged with GLOBAL_LOAD_ASYNC_TO_LDS_B128 (ASYNCcnt) when available
// ---------------------------------------------------------------------------

typedef __attribute__((ext_vector_type(16))) __bf16 v16bf;
typedef __attribute__((ext_vector_type(8)))  float  v8f;

// async-to-LDS builtin operand types (from hipcc diagnostic: int4 pointers)
typedef int v4i __attribute__((vector_size(16)));
typedef __attribute__((address_space(1))) v4i v4i_g;
typedef __attribute__((address_space(3))) v4i v4i_l;

#define BM 128
#define BN 128
#define BK 32
#define SPAD 8   // LDS row padding (bf16 elems) -> row stride 80B (16B aligned)

#if defined(__has_builtin)
#  if __has_builtin(__builtin_amdgcn_global_load_async_to_lds_b128)
#    define HAVE_ASYNC_LDS 1
#  endif
#endif
#ifndef HAVE_ASYNC_LDS
#  define HAVE_ASYNC_LDS 0
#endif

__device__ __forceinline__ void async_copy16_g2l(const void* gp, void* lp)
{
#if HAVE_ASYNC_LDS
    __builtin_amdgcn_global_load_async_to_lds_b128(
        (v4i_g*)(uintptr_t)gp, (v4i_l*)lp, 0, 0);
#else
    *(uint4*)lp = *(const uint4*)gp;
#endif
}

__device__ __forceinline__ void wait_async_zero()
{
#if HAVE_ASYNC_LDS
#  if __has_builtin(__builtin_amdgcn_s_wait_asynccnt)
    __builtin_amdgcn_s_wait_asynccnt(0);
#  else
    asm volatile("s_wait_asynccnt 0x0" ::: "memory");
#  endif
#endif
}

// ---------------------------------------------------------------------------
// Generic batched GEMM:  C = alpha * A * B (+ C if accum)
//   A[m,k] = A  + m*lda + k                  (fp32, k stride 1)
//   bBf16==0:  B[k,n] = Bf + k*ldb + n*stB   (fp32)
//   bBf16==1:  B[k,n] = Bh + n*stB + k       (bf16, k-contiguous; async path)
//   C[m,n] = C + m*ldc + n                   (fp32)
// blockIdx.z = bb*Hdiv + hh ; offsets bb*sXb + hh*sXh address (batch, head)
// sub-matrices inside interleaved [B,T,heads*dk] tensors.
// ---------------------------------------------------------------------------
__global__ __launch_bounds__(256)
void hrm_gemm_wmma(const float* __restrict__ A, const void* __restrict__ Bv,
                   float* __restrict__ C,
                   int M, int N, int K,
                   long lda, long ldb, long stB, long ldc,
                   long sAb, long sAh, long sBb, long sBh, long sCb, long sCh,
                   int Hdiv, float alpha, int accum, int bBf16)
{
    const int z  = blockIdx.z;
    const int bb = z / Hdiv;
    const int hh = z - bb * Hdiv;
    A += (long)bb * sAb + (long)hh * sAh;
    C += (long)bb * sCb + (long)hh * sCh;
    const float*  Bf = 0;
    const __bf16* Bh = 0;
    if (bBf16) Bh = (const __bf16*)Bv + (long)bb * sBb + (long)hh * sBh;
    else       Bf = (const float*)Bv + (long)bb * sBb + (long)hh * sBh;

    __shared__ __bf16 As[BM][BK + SPAD];  // [m][k]
    __shared__ __bf16 Bs[BN][BK + SPAD];  // [n][k]  (k-contiguous frag loads)

    const int tid      = threadIdx.x;
    const int wave     = tid >> 5;
    const int lane     = tid & 31;
    const int waveM    = wave >> 1;     // 0..3 : 32 rows each
    const int waveN    = wave & 1;      // 0..1 : 64 cols each
    const int lane15   = lane & 15;
    const int laneHalf = lane >> 4;

    const int m0 = blockIdx.y * BM;
    const int n0 = blockIdx.x * BN;

    v8f acc[2][4];
    const v8f vzero = {0.f, 0.f, 0.f, 0.f, 0.f, 0.f, 0.f, 0.f};
#pragma unroll
    for (int i = 0; i < 2; ++i)
#pragma unroll
        for (int j = 0; j < 4; ++j) acc[i][j] = vzero;

    for (int k0 = 0; k0 < K; k0 += BK) {
        // ---- stage B tile first (async path runs while we stage A) ----
        if (bBf16) {
            // 16B chunks: 8 bf16 along k
            for (int c = tid; c < (BN * BK) / 8; c += 256) {
                int n  = c >> 2;
                int kc = (c & 3) << 3;
                int gn = n0 + n;
                if (gn < N) {
                    async_copy16_g2l(Bh + (long)gn * stB + (k0 + kc), &Bs[n][kc]);
                } else {
                    uint4 zz = {0u, 0u, 0u, 0u};
                    *(uint4*)&Bs[n][kc] = zz;
                }
            }
        } else if (ldb == 1) {
            // fp32, k-contiguous (e.g. q.k^T B operand): float4 + convert
            for (int c = tid; c < (BN * BK) / 4; c += 256) {
                int n  = c >> 3;
                int kc = (c & 7) << 2;
                int gn = n0 + n, gk = k0 + kc;
                union { __bf16 h[4]; uint2 u2; } pk;
                if (gn < N && gk + 3 < K) {
                    float4 v = *(const float4*)(Bf + (long)gn * stB + gk);
                    pk.h[0] = (__bf16)v.x; pk.h[1] = (__bf16)v.y;
                    pk.h[2] = (__bf16)v.z; pk.h[3] = (__bf16)v.w;
                } else {
                    for (int e = 0; e < 4; ++e) {
                        float vv = 0.f;
                        if (gn < N && gk + e < K) vv = Bf[(long)gn * stB + gk + e];
                        pk.h[e] = (__bf16)vv;
                    }
                }
                *(uint2*)&Bs[n][kc] = pk.u2;
            }
        } else {
            // fp32, generic strides (softmax(S).V operand): scalar
            for (int i = tid; i < BK * BN; i += 256) {
                int kk = i >> 7;
                int n  = i & (BN - 1);
                int gn = n0 + n, gk = k0 + kk;
                float v = 0.f;
                if (gn < N && gk < K) v = Bf[(long)gk * ldb + (long)gn * stB];
                Bs[n][kk] = (__bf16)v;
            }
        }

        // ---- stage A tile (BM x BK): float4 loads + packed bf16 stores ----
        for (int c = tid; c < (BM * BK) / 4; c += 256) {
            int m  = c >> 3;
            int kc = (c & 7) << 2;
            int gm = m0 + m, gk = k0 + kc;
            union { __bf16 h[4]; uint2 u2; } pk;
            if (gm < M && gk + 3 < K) {
                float4 v = *(const float4*)(A + (long)gm * lda + gk);
                pk.h[0] = (__bf16)v.x; pk.h[1] = (__bf16)v.y;
                pk.h[2] = (__bf16)v.z; pk.h[3] = (__bf16)v.w;
            } else {
                for (int e = 0; e < 4; ++e) {
                    float vv = 0.f;
                    if (gm < M && gk + e < K) vv = A[(long)gm * lda + gk + e];
                    pk.h[e] = (__bf16)vv;
                }
            }
            *(uint2*)&As[m][kc] = pk.u2;
        }

        // prefetch next A tile rows (global_prefetch_b8)
        if (k0 + BK < K) {
            int gm = m0 + (tid >> 1);
            if (gm < M)
                __builtin_prefetch(&A[(long)gm * lda + k0 + BK + (tid & 1) * 16], 0, 1);
        }

        if (bBf16) wait_async_zero();
        __syncthreads();

        // ---- fragments (ISA 16-bit A 16x32 layout; B symmetric) ----
        v16bf afrag[2];
#pragma unroll
        for (int t = 0; t < 2; ++t) {
            const int mr = waveM * 32 + t * 16 + lane15;
#pragma unroll
            for (int j = 0; j < 16; ++j) {
                const int kk = (j < 8) ? (8 * laneHalf + j)
                                       : (16 + 8 * laneHalf + (j - 8));
                afrag[t][j] = As[mr][kk];
            }
        }
        v16bf bfrag[4];
#pragma unroll
        for (int t = 0; t < 4; ++t) {
            const int nc = waveN * 64 + t * 16 + lane15;
#pragma unroll
            for (int j = 0; j < 16; ++j) {
                const int kk = (j < 8) ? (8 * laneHalf + j)
                                       : (16 + 8 * laneHalf + (j - 8));
                bfrag[t][j] = Bs[nc][kk];
            }
        }

#pragma unroll
        for (int i = 0; i < 2; ++i)
#pragma unroll
            for (int j = 0; j < 4; ++j)
                acc[i][j] = __builtin_amdgcn_wmma_f32_16x16x32_bf16(
                    false, afrag[i], false, bfrag[j], (short)0, acc[i][j],
                    false, false);

        __syncthreads();
    }

    // ---- writeback: VGPR r -> M = r + 8*laneHalf ; N = lane15 ----
#pragma unroll
    for (int i = 0; i < 2; ++i) {
#pragma unroll
        for (int j = 0; j < 4; ++j) {
            const int nc = n0 + waveN * 64 + j * 16 + lane15;
            if (nc >= N) continue;
#pragma unroll
            for (int r = 0; r < 8; ++r) {
                const int mr = m0 + waveM * 32 + i * 16 + 8 * laneHalf + r;
                if (mr < M) {
                    long off = (long)mr * ldc + nc;
                    float v = alpha * acc[i][j][r];
                    if (accum) v += C[off];
                    C[off] = v;
                }
            }
        }
    }
}

// ---------------------------------------------------------------------------
// Weight fp32 [K,N] -> bf16 [N,K] (k-contiguous), once per call
// ---------------------------------------------------------------------------
__global__ void convert_transpose_kernel(const float* __restrict__ W,
                                         __bf16* __restrict__ Wt, int K, int N)
{
    long idx = (long)blockIdx.x * blockDim.x + threadIdx.x;
    if (idx >= (long)K * N) return;
    long n = idx / K;
    int  k = (int)(idx - n * K);
    Wt[idx] = (__bf16)W[(long)k * N + n];
}

// ---------------------------------------------------------------------------
// Elementwise / reduction kernels (blockDim = 256)
// ---------------------------------------------------------------------------
__global__ void rmsnorm_kernel(const float* __restrict__ x,
                               const float* __restrict__ w,
                               float* __restrict__ y, int D)
{
    __shared__ float red[256];
    const long row = blockIdx.x;
    const float* xr = x + row * D;
    float* yr = y + row * D;
    const int tid = threadIdx.x;
    float s = 0.f;
    for (int i = tid; i < D; i += 256) { float v = xr[i]; s += v * v; }
    red[tid] = s; __syncthreads();
    for (int o = 128; o > 0; o >>= 1) {
        if (tid < o) red[tid] += red[tid + o];
        __syncthreads();
    }
    const float inv = rsqrtf(red[0] / (float)D + 1e-5f);
    for (int i = tid; i < D; i += 256) yr[i] = w[i] * xr[i] * inv;
}

// RoPE over full dim (pre head-split) applied in-place to q,k of [B*T, 3D]
__global__ void rope_kernel(float* qkv, int Bn, int T, int D)
{
    long idx = (long)blockIdx.x * blockDim.x + threadIdx.x;
    const int half = D >> 1;
    const long total = (long)Bn * T * D;
    if (idx >= total) return;
    const int  p   = (int)(idx % half);
    long r = idx / half;
    const int  qk  = (int)(r & 1);
    const long row = r >> 1;
    const int  t   = (int)(row % T);
    const float ang = (float)t * __powf(10000.f, -(float)p / (float)half);
    const float c = __cosf(ang), s = __sinf(ang);
    float* base = qkv + row * (3L * D) + (long)qk * D + 2 * p;
    const float xr = base[0], xi = base[1];
    base[0] = xr * c - xi * s;
    base[1] = xr * s + xi * c;
}

__global__ void softmax_kernel(float* x, int L)
{
    __shared__ float red[256];
    float* row = x + (long)blockIdx.x * L;
    const int tid = threadIdx.x;
    float m = -3.0e38f;
    for (int i = tid; i < L; i += 256) m = fmaxf(m, row[i]);
    red[tid] = m; __syncthreads();
    for (int o = 128; o > 0; o >>= 1) {
        if (tid < o) red[tid] = fmaxf(red[tid], red[tid + o]);
        __syncthreads();
    }
    m = red[0]; __syncthreads();
    float s = 0.f;
    for (int i = tid; i < L; i += 256) {
        float e = __expf(row[i] - m);
        row[i] = e; s += e;
    }
    red[tid] = s; __syncthreads();
    for (int o = 128; o > 0; o >>= 1) {
        if (tid < o) red[tid] += red[tid + o];
        __syncthreads();
    }
    const float inv = 1.f / red[0];
    for (int i = tid; i < L; i += 256) row[i] *= inv;
}

__global__ void glu_kernel(const float* __restrict__ fc, float* __restrict__ out,
                           long Mrows, int inner)
{
    long idx = (long)blockIdx.x * blockDim.x + threadIdx.x;
    if (idx >= Mrows * (long)inner) return;
    const long m = idx / inner;
    const int  j = (int)(idx - m * inner);
    const float a = fc[m * 2L * inner + j];
    const float g = fc[m * 2L * inner + inner + j];
    out[idx] = a * (1.f / (1.f + __expf(-g)));
}

__global__ void pool_kernel(const float* __restrict__ x, float* __restrict__ p,
                            int Bn, int T, int D)
{
    const int idx = blockIdx.x * blockDim.x + threadIdx.x;
    if (idx >= Bn * D) return;
    const int b = idx / D, d = idx - b * D;
    const float* base = x + (long)b * T * D + d;
    float s = 0.f;
    for (int t = 0; t < T; ++t) s += base[(long)t * D];
    p[idx] = s / (float)T;
}

__global__ void film_kernel(float* l, const float* __restrict__ g,
                            const float* __restrict__ bv, int Bn, int T, int D)
{
    long idx = (long)blockIdx.x * blockDim.x + threadIdx.x;
    const long tot = (long)Bn * T * D;
    if (idx >= tot) return;
    const int  d  = (int)(idx % D);
    const long bt = idx / D;
    const int  b  = (int)(bt / T);
    l[idx] = l[idx] * (1.f + g[b * D + d]) + bv[b * D + d];
}

__global__ void heads_kernel(const float* __restrict__ hp, const float* __restrict__ lp,
                             const float* __restrict__ wa, const float* __restrict__ wb,
                             const float* __restrict__ wq,
                             float* outA, float* outB, float* qlog,
                             int HDk, int LDk)
{
    __shared__ float red[256];
    const int b = blockIdx.x;
    const int tid = threadIdx.x;

    float s = 0.f;
    for (int i = tid; i < HDk; i += 256) s += hp[b * HDk + i] * wa[i];
    red[tid] = s; __syncthreads();
    for (int o = 128; o > 0; o >>= 1) { if (tid < o) red[tid] += red[tid + o]; __syncthreads(); }
    if (tid == 0) outA[b] = red[0];
    __syncthreads();

    s = 0.f;
    for (int i = tid; i < LDk; i += 256) s += lp[b * LDk + i] * wb[i];
    red[tid] = s; __syncthreads();
    for (int o = 128; o > 0; o >>= 1) { if (tid < o) red[tid] += red[tid + o]; __syncthreads(); }
    if (tid == 0) outB[b] = red[0];
    __syncthreads();

    for (int j = 0; j < 2; ++j) {
        s = 0.f;
        for (int i = tid; i < HDk; i += 256) s += hp[b * HDk + i] * wq[i * 2 + j];
        red[tid] = s; __syncthreads();
        for (int o = 128; o > 0; o >>= 1) { if (tid < o) red[tid] += red[tid + o]; __syncthreads(); }
        if (tid == 0) qlog[b * 2 + j] = red[0];
        __syncthreads();
    }
}

// ---------------------------------------------------------------------------
// Host-side orchestration
// ---------------------------------------------------------------------------
static inline void launch_gemm(hipStream_t s,
                               const float* A, const void* B, float* C,
                               int M, int N, int K,
                               long lda, long ldb, long stB, long ldc,
                               long sAb, long sAh, long sBb, long sBh,
                               long sCb, long sCh,
                               int nbatch, int Hdiv, float alpha, int accum,
                               int bBf16)
{
    dim3 grid((N + BN - 1) / BN, (M + BM - 1) / BM, nbatch);
    hrm_gemm_wmma<<<grid, dim3(256), 0, s>>>(A, B, C, M, N, K,
                                             lda, ldb, stB, ldc,
                                             sAb, sAh, sBb, sBh, sCb, sCh,
                                             Hdiv, alpha, accum, bBf16);
}

// standard dense layer GEMM: A [M,K] fp32 contiguous, Wt bf16 [N,K], C [M,N]
static inline void gemm_w(hipStream_t s, const float* A, const __bf16* Wt,
                          float* C, int M, int N, int K, float alpha, int accum)
{
    launch_gemm(s, A, Wt, C, M, N, K,
                K, 1, K, N, 0, 0, 0, 0, 0, 0, 1, 1, alpha, accum, 1);
}

struct LayerW {
    const __bf16 *attn_proj, *fc, *ffn_proj, *qkv;
    const float  *n1, *n2;
};

static void encoder_layer(hipStream_t s, float* x, const LayerW& w,
                          int Bn, int T, int D, int inner, int heads,
                          float* nbuf, float* qkvb, float* scores,
                          float* atto, float* fcb, float* actb)
{
    const int  Mrows = Bn * T;
    const int  dk    = D / heads;
    const long ld3   = 3L * D;

    // ---- attention ----
    rmsnorm_kernel<<<Mrows, 256, 0, s>>>(x, w.n1, nbuf, D);
    gemm_w(s, nbuf, w.qkv, qkvb, Mrows, 3 * D, D, 1.f, 0);
    long nrope = (long)Bn * T * D;
    rope_kernel<<<(unsigned)((nrope + 255) / 256), 256, 0, s>>>(qkvb, Bn, T, D);

    // S = q . k^T / sqrt(dk), per (b,h)
    launch_gemm(s, qkvb /*q*/, qkvb + D /*k*/, scores,
                T, T, dk,
                ld3, 1, ld3, T,
                (long)T * 3 * D, dk, (long)T * 3 * D, dk,
                (long)heads * T * T, (long)T * T,
                Bn * heads, heads, 1.f / sqrtf((float)dk), 0, 0);
    softmax_kernel<<<Bn * heads * T, 256, 0, s>>>(scores, T);
    // O = S . v
    launch_gemm(s, scores, qkvb + 2 * D /*v*/, atto,
                T, dk, T,
                T, ld3, 1, D,
                (long)heads * T * T, (long)T * T, (long)T * 3 * D, dk,
                (long)T * D, dk,
                Bn * heads, heads, 1.f, 0, 0);
    // x += O @ attn_proj
    gemm_w(s, atto, w.attn_proj, x, Mrows, D, D, 1.f, 1);

    // ---- GLU FFN ----
    rmsnorm_kernel<<<Mrows, 256, 0, s>>>(x, w.n2, nbuf, D);
    gemm_w(s, nbuf, w.fc, fcb, Mrows, 2 * inner, D, 1.f, 0);
    long nglu = (long)Mrows * inner;
    glu_kernel<<<(unsigned)((nglu + 255) / 256), 256, 0, s>>>(fcb, actb, Mrows, inner);
    gemm_w(s, actb, w.ffn_proj, x, Mrows, D, inner, 1.f, 1);
}

extern "C" void kernel_launch(void* const* d_in, const int* in_sizes, int n_in,
                              void* d_out, int out_size, void* d_ws, size_t ws_size,
                              hipStream_t stream)
{
    (void)in_sizes; (void)n_in; (void)out_size; (void)ws_size;

    const int B = 4, TH = 256, TL = 512;
    const int HDIM = 512, LDIM = 768;
    const int HINNER = 1536, LINNER = 2304;
    const int HHEADS = 8, LHEADS = 12;

    // --- input leaves, jax pytree order (dict keys sorted) ---
    // 0 h_tokens, 1 l_tokens,
    // 2 cross.wk, 3 cross.wo, 4 cross.wq, 5 cross.wv, 6 film_b, 7 film_g,
    // 8+6i h_enc[i]: attn_proj, fc, ffn_proj, n1, n2, qkv   (i=0..3)
    // 32 h_norm, 33 headA, 34 headB,
    // 35+6i l_enc[i]: attn_proj, fc, ffn_proj, n1, n2, qkv  (i=0..5)
    // 71 l_norm, 72 q_head
    auto F = [&](int i) -> const float* { return (const float*)d_in[i]; };
    const float* h_tokens = F(0);
    const float* l_tokens = F(1);
    const float* h_norm = F(32);
    const float* headA  = F(33);
    const float* headB  = F(34);
    const float* l_norm = F(71);
    const float* q_head = F(72);

    // --- weight descriptor table: leaf index, K, N ---
    struct WDesc { int idx, K, N; };
    WDesc wd[64];
    int nw = 0;
    wd[nw++] = {2, HDIM, LDIM};          // cross.wk
    wd[nw++] = {3, LDIM, LDIM};          // cross.wo
    wd[nw++] = {4, LDIM, LDIM};          // cross.wq
    wd[nw++] = {5, HDIM, LDIM};          // cross.wv
    wd[nw++] = {6, HDIM, LDIM};          // film_b
    wd[nw++] = {7, HDIM, LDIM};          // film_g
    for (int i = 0; i < 4; ++i) {
        int base = 8 + 6 * i;
        wd[nw++] = {base + 0, HDIM, HDIM};        // attn_proj
        wd[nw++] = {base + 1, HDIM, 2 * HINNER};  // fc
        wd[nw++] = {base + 2, HINNER, HDIM};      // ffn_proj
        wd[nw++] = {base + 5, HDIM, 3 * HDIM};    // qkv
    }
    for (int i = 0; i < 6; ++i) {
        int base = 35 + 6 * i;
        wd[nw++] = {base + 0, LDIM, LDIM};
        wd[nw++] = {base + 1, LDIM, 2 * LINNER};
        wd[nw++] = {base + 2, LINNER, LDIM};
        wd[nw++] = {base + 5, LDIM, 3 * LDIM};
    }

    // --- workspace carve ---
    float* w = (float*)d_ws;
    long off = 0;
    auto alloc = [&](long n) { float* p = w + off; off += n; return p; };
    const long NH = (long)B * TH * HDIM;   // 524288
    const long NL = (long)B * TL * LDIM;   // 1572864
    float* Hbuf = alloc(NH);
    float* Lbuf = alloc(NL);
    float* Nbuf = alloc(NL);
    float* qkvb = alloc((long)B * TL * 3 * LDIM);
    // union: scores (<= B*12*512*512) OR fc (B*TL*2*LINNER) + act
    float* R    = alloc((long)B * TL * 2 * LINNER + (long)B * TL * LINNER);
    float* scores = R;
    float* fcb    = R;
    float* actb   = R + (long)B * TL * 2 * LINNER;
    float* atto  = alloc(NL);
    float* qb    = alloc(NL);
    float* kb    = alloc((long)B * TH * LDIM);
    float* vb    = alloc((long)B * TH * LDIM);
    float* hpool = alloc(B * HDIM);
    float* lpool = alloc(B * LDIM);
    float* gvec  = alloc(B * LDIM);
    float* bvec  = alloc(B * LDIM);

    // bf16 weight area (16B aligned)
    off = (off + 3) & ~3L;
    __bf16* wtb = (__bf16*)(w + off);
    __bf16* wt[80];
    long boff = 0;
    for (int i = 0; i < nw; ++i) {
        wt[wd[i].idx] = wtb + boff;
        boff += (long)wd[i].K * wd[i].N;
    }

    // --- convert + transpose all weights to bf16 [N,K] ---
    for (int i = 0; i < nw; ++i) {
        long tot = (long)wd[i].K * wd[i].N;
        convert_transpose_kernel<<<(unsigned)((tot + 255) / 256), 256, 0, stream>>>(
            F(wd[i].idx), wt[wd[i].idx], wd[i].K, wd[i].N);
    }

    (void)hipMemcpyAsync(Hbuf, h_tokens, sizeof(float) * NH, hipMemcpyDeviceToDevice, stream);
    (void)hipMemcpyAsync(Lbuf, l_tokens, sizeof(float) * NL, hipMemcpyDeviceToDevice, stream);

    float* outp = (float*)d_out;
    float* outA = outp + NH + NL;
    float* outB = outA + B;
    float* qlog = outB + B;

    for (int step = 0; step < 4; ++step) {
        // ---- H encoder ----
        for (int li = 0; li < 4; ++li) {
            const int base = 8 + 6 * li;
            LayerW lw{wt[base + 0], wt[base + 1], wt[base + 2], wt[base + 5],
                      F(base + 3), F(base + 4)};
            encoder_layer(stream, Hbuf, lw, B, TH, HDIM, HINNER, HHEADS,
                          Nbuf, qkvb, scores, atto, fcb, actb);
        }
        // ---- h_pool = mean(rmsnorm(h)) ----
        rmsnorm_kernel<<<B * TH, 256, 0, stream>>>(Hbuf, h_norm, Nbuf, HDIM);
        pool_kernel<<<(B * HDIM + 255) / 256, 256, 0, stream>>>(Nbuf, hpool, B, TH, HDIM);

        // ---- FiLM conditioning of L ----
        gemm_w(stream, hpool, wt[7], gvec, B, LDIM, HDIM, 1.f, 0);  // film_g
        gemm_w(stream, hpool, wt[6], bvec, B, LDIM, HDIM, 1.f, 0);  // film_b
        long nfilm = (long)B * TL * LDIM;
        film_kernel<<<(unsigned)((nfilm + 255) / 256), 256, 0, stream>>>(Lbuf, gvec, bvec, B, TL, LDIM);

        // ---- cross attention: cond += CA(cond, h) ----
        gemm_w(stream, Lbuf, wt[4], qb, B * TL, LDIM, LDIM, 1.f, 0);  // wq
        gemm_w(stream, Hbuf, wt[2], kb, B * TH, LDIM, HDIM, 1.f, 0);  // wk
        gemm_w(stream, Hbuf, wt[5], vb, B * TH, LDIM, HDIM, 1.f, 0);  // wv
        const int cdk = LDIM / LHEADS;  // 64
        launch_gemm(stream, qb, kb, scores,
                    TL, TH, cdk,
                    LDIM, 1, LDIM, TH,
                    (long)TL * LDIM, cdk, (long)TH * LDIM, cdk,
                    (long)LHEADS * TL * TH, (long)TL * TH,
                    B * LHEADS, LHEADS, 1.f / sqrtf((float)cdk), 0, 0);
        softmax_kernel<<<B * LHEADS * TL, 256, 0, stream>>>(scores, TH);
        launch_gemm(stream, scores, vb, atto,
                    TL, cdk, TH,
                    TH, LDIM, 1, LDIM,
                    (long)LHEADS * TL * TH, (long)TL * TH, (long)TH * LDIM, cdk,
                    (long)TL * LDIM, cdk,
                    B * LHEADS, LHEADS, 1.f, 0, 0);
        gemm_w(stream, atto, wt[3], Lbuf, B * TL, LDIM, LDIM, 1.f, 1);  // wo, +=

        // ---- L encoder ----
        for (int li = 0; li < 6; ++li) {
            const int base = 35 + 6 * li;
            LayerW lw{wt[base + 0], wt[base + 1], wt[base + 2], wt[base + 5],
                      F(base + 3), F(base + 4)};
            encoder_layer(stream, Lbuf, lw, B, TL, LDIM, LINNER, LHEADS,
                          Nbuf, qkvb, scores, atto, fcb, actb);
        }
        // ---- l_pool + heads (overwritten each step; last one counts) ----
        rmsnorm_kernel<<<B * TL, 256, 0, stream>>>(Lbuf, l_norm, Nbuf, LDIM);
        pool_kernel<<<(B * LDIM + 255) / 256, 256, 0, stream>>>(Nbuf, lpool, B, TL, LDIM);
        heads_kernel<<<B, 256, 0, stream>>>(hpool, lpool, headA, headB, q_head,
                                            outA, outB, qlog, HDIM, LDIM);
    }

    // ---- final outputs: (h, l, outA, outB, q_logits) concatenated ----
    (void)hipMemcpyAsync(outp, Hbuf, sizeof(float) * NH, hipMemcpyDeviceToDevice, stream);
    (void)hipMemcpyAsync(outp + NH, Lbuf, sizeof(float) * NL, hipMemcpyDeviceToDevice, stream);
}
